// MultiHeadSelfAttention_58102317580386
// MI455X (gfx1250) — matrix operand order repaired and checked by reading, not compiled, not run on hardware
//
#include <hip/hip_runtime.h>

typedef __bf16 bf16;
typedef __attribute__((ext_vector_type(16))) __bf16 v16bf;
typedef __attribute__((ext_vector_type(8)))  float  v8f;
typedef __attribute__((ext_vector_type(4)))  unsigned int v4u;

#define EMBED    1024
#define HEADS    16
#define HEAD_DIM 64
#define BATCH    4
#define SEQ      2048
#define ROWS     (BATCH * SEQ)          // 8192

// ---------------------------------------------------------------------------
// Cross-lane reductions within each 16-lane row via v_permlane16_b32 (VALU,
// no LDS round-trip). Selector nibbles implement XOR-1/2/4/8 butterflies.
// ---------------------------------------------------------------------------
__device__ __forceinline__ float pl16(float v, unsigned lo, unsigned hi) {
    int s = __float_as_int(v);
    int r = __builtin_amdgcn_permlane16(s, s, lo, hi, false, false);
    return __int_as_float(r);
}
__device__ __forceinline__ float rowmax16(float v) {
    v = fmaxf(v, pl16(v, 0x67452301u, 0xEFCDAB89u));  // xor 1
    v = fmaxf(v, pl16(v, 0x54761032u, 0xDCFE98BAu));  // xor 2
    v = fmaxf(v, pl16(v, 0x32107654u, 0xBA98FEDCu));  // xor 4
    v = fmaxf(v, pl16(v, 0xFEDCBA98u, 0x76543210u));  // xor 8
    return v;
}
__device__ __forceinline__ float rowsum16(float v) {
    v += pl16(v, 0x67452301u, 0xEFCDAB89u);
    v += pl16(v, 0x54761032u, 0xDCFE98BAu);
    v += pl16(v, 0x32107654u, 0xBA98FEDCu);
    v += pl16(v, 0xFEDCBA98u, 0x76543210u);
    return v;
}

// ---------------------------------------------------------------------------
// Weight convert + transpose: Wt[k][j] = (bf16) W[j][k]
// ---------------------------------------------------------------------------
__global__ void wt_convert_kernel(const float* __restrict__ W, bf16* __restrict__ Wt) {
    int idx = blockIdx.x * 256 + threadIdx.x;      // j*1024 + k
    int k = idx & (EMBED - 1);
    int j = idx >> 10;
    Wt[(size_t)k * EMBED + j] = (bf16)W[idx];
}

// ---------------------------------------------------------------------------
// 64x128-tile GEMM: Out = A @ Wt (+bias).  M=8192, N=1024, K=1024.
// 128 threads = 4 waves; wave w computes rows 16w..16w+15 x 128 cols (8 tiles).
// ---------------------------------------------------------------------------
template <bool A_IS_F32, bool FINAL>
__global__ void __launch_bounds__(128, 1)
gemm64_kernel(const void* __restrict__ Aptr,
              const bf16* __restrict__ Bt,
              const float* __restrict__ bias,
              void* __restrict__ Out) {
    __shared__ __align__(16) bf16 As[64][40];    // [m][k]  row stride 80B
    __shared__ __align__(16) bf16 Bs[128][40];   // [n][k]

    const int tid  = threadIdx.x;
    const int lane = tid & 31;
    const int wave = tid >> 5;
    const int kh   = lane >> 4;      // 0/1: 16-lane half
    const int ln   = lane & 15;
    const int m0   = blockIdx.x * 64;
    const int n0   = blockIdx.y * 128;

    v8f acc[8] = {};

    for (int kc = 0; kc < EMBED; kc += 32) {
        __syncthreads();
        // ---- stage A tile (64 x 32) ----
        if (A_IS_F32) {
            const float* A = (const float*)Aptr;
#pragma unroll
            for (int i = 0; i < 4; ++i) {
                int idx = tid + 128 * i;          // 512 groups of 4 floats
                int r = idx >> 3, g = idx & 7;
                float4 f = *(const float4*)(A + (size_t)(m0 + r) * EMBED + kc + 4 * g);
                bf16* d = &As[r][4 * g];
                d[0] = (bf16)f.x; d[1] = (bf16)f.y; d[2] = (bf16)f.z; d[3] = (bf16)f.w;
            }
            if (kc + 32 < EMBED)
                __builtin_prefetch((const char*)A + ((size_t)(m0 + (tid >> 1)) * EMBED + kc + 32) * 4, 0, 0);
        } else {
            const bf16* A = (const bf16*)Aptr;
#pragma unroll
            for (int i = 0; i < 2; ++i) {
                int idx = tid + 128 * i;          // 256 groups of 8 halves
                int r = idx >> 2, g = idx & 3;
                *(v4u*)&As[r][8 * g] = *(const v4u*)(A + (size_t)(m0 + r) * EMBED + kc + 8 * g);
            }
            if (kc + 32 < EMBED)
                __builtin_prefetch((const char*)A + ((size_t)(m0 + (tid >> 1)) * EMBED + kc + 32) * 2, 0, 0);
        }
        // ---- stage B tile (32 k x 128 n) transposed into [n][k] ----
#pragma unroll
        for (int i = 0; i < 4; ++i) {
            int idx = tid + 128 * i;              // 512 groups of 8 halves
            int kl = idx >> 4, g = idx & 15;
            union { v4u u; bf16 e[8]; } t;
            t.u = *(const v4u*)(Bt + (size_t)(kc + kl) * EMBED + n0 + 8 * g);
#pragma unroll
            for (int e = 0; e < 8; ++e) Bs[8 * g + e][kl] = t.e[e];
        }
        __syncthreads();

        // ---- fragments + WMMA ----
        union { v16bf v; v4u q[2]; } af;
        af.q[0] = *(const v4u*)&As[16 * wave + ln][8 * kh];        // K = h+8kh,  h<8
        af.q[1] = *(const v4u*)&As[16 * wave + ln][16 + 8 * kh];   // K = h+8+8kh
#pragma unroll
        for (int t = 0; t < 8; ++t) {
            union { v16bf v; v4u q[2]; } bf;
            int n = 16 * t + ln;
            bf.q[0] = *(const v4u*)&Bs[n][8 * kh];                 // K = h+8kh, h<8
            bf.q[1] = *(const v4u*)&Bs[n][16 + 8 * kh];
            acc[t] = __builtin_amdgcn_wmma_f32_16x16x32_bf16(
                false, af.v, false, bf.v, (short)0, acc[t], false, false);
        }
    }

    // ---- epilogue ----
    if (FINAL) {
        float* O = (float*)Out;
#pragma unroll
        for (int t = 0; t < 8; ++t) {
            int col = n0 + 16 * t + ln;
            float bb = bias[col];
#pragma unroll
            for (int r = 0; r < 8; ++r) {
                int row = m0 + 16 * wave + r + 8 * kh;
                O[(size_t)row * EMBED + col] = acc[t][r] + bb;
            }
        }
    } else {
        bf16* O = (bf16*)Out;                                       // [B,H,S,64]
#pragma unroll
        for (int t = 0; t < 8; ++t) {
            int col = n0 + 16 * t + ln;
            int hh = col >> 6, dh = col & 63;
#pragma unroll
            for (int r = 0; r < 8; ++r) {
                int row = m0 + 16 * wave + r + 8 * kh;              // b*2048+s
                int b = row >> 11, s = row & (SEQ - 1);
                O[(((size_t)(b * HEADS + hh)) * SEQ + s) * HEAD_DIM + dh] = (bf16)acc[t][r];
            }
        }
    }
}

// ---------------------------------------------------------------------------
// Flash attention: one workgroup = (b, h, 64 query rows); 4 waves x 16 rows.
// Online softmax with scale 1/sqrt(EMBED)=1/32 (matches reference).
// ---------------------------------------------------------------------------
__global__ void __launch_bounds__(128, 1)
flash_attn_kernel(const bf16* __restrict__ Qp,
                  const bf16* __restrict__ Kp,
                  const bf16* __restrict__ Vp,
                  bf16* __restrict__ Out) {
    __shared__ __align__(16) bf16 Ks[64][72];      // [key][d]
    __shared__ __align__(16) bf16 Vs[64][72];      // [d][key]  (transposed)
    __shared__ __align__(16) bf16 Ps[4][16][72];   // per-wave P staging

    const int tid  = threadIdx.x;
    const int lane = tid & 31;
    const int wave = tid >> 5;
    const int kh   = lane >> 4;
    const int ln   = lane & 15;
    const int qb   = blockIdx.x;     // 0..31
    const int h    = blockIdx.y;     // 0..15
    const int b    = blockIdx.z;     // 0..3

    const size_t headoff = ((size_t)(b * HEADS + h)) * SEQ * HEAD_DIM;
    const bf16* Qh = Qp + headoff;
    const bf16* Kh = Kp + headoff;
    const bf16* Vh = Vp + headoff;
    const int qrow0 = qb * 64 + 16 * wave;

    // Q fragments (16 rows x 64 d -> 2 chunks of K=32), loaded once
    v16bf qf[2];
#pragma unroll
    for (int c = 0; c < 2; ++c) {
        union { v16bf v; v4u q[2]; } u;
        const bf16* base = Qh + (size_t)(qrow0 + ln) * HEAD_DIM + 32 * c;
        u.q[0] = *(const v4u*)(base + 8 * kh);
        u.q[1] = *(const v4u*)(base + 16 + 8 * kh);
        qf[c] = u.v;
    }

    float m_r[8], l_r[8];
#pragma unroll
    for (int r = 0; r < 8; ++r) { m_r[r] = -3.0e38f; l_r[r] = 0.0f; }
    v8f acc_o[4] = {};

    for (int kb = 0; kb < SEQ / 64; ++kb) {
        __syncthreads();
        // ---- stage K tile [key][d] ----
#pragma unroll
        for (int i = 0; i < 4; ++i) {
            int idx = tid + 128 * i;                 // 512 groups of 8 halves
            int r = idx >> 3, g = idx & 7;
            *(v4u*)&Ks[r][8 * g] = *(const v4u*)(Kh + (size_t)(kb * 64 + r) * HEAD_DIM + 8 * g);
        }
        // ---- stage V tile transposed [d][key] ----
#pragma unroll
        for (int i = 0; i < 4; ++i) {
            int idx = tid + 128 * i;
            int r = idx >> 3, g = idx & 7;           // r=key, g=d-group
            union { v4u u; bf16 e[8]; } vv;
            vv.u = *(const v4u*)(Vh + (size_t)(kb * 64 + r) * HEAD_DIM + 8 * g);
#pragma unroll
            for (int e = 0; e < 8; ++e) Vs[8 * g + e][r] = vv.e[e];
        }
        __syncthreads();

        // ---- S = Q K^T  (16 rows x 64 keys) ----
        v8f s[4] = {};
#pragma unroll
        for (int c = 0; c < 2; ++c) {
#pragma unroll
            for (int t = 0; t < 4; ++t) {
                union { v16bf v; v4u q[2]; } kf;     // B: [d][key16]
                int key = 16 * t + ln;
                kf.q[0] = *(const v4u*)&Ks[key][32 * c + 8 * kh];
                kf.q[1] = *(const v4u*)&Ks[key][32 * c + 16 + 8 * kh];
                s[t] = __builtin_amdgcn_wmma_f32_16x16x32_bf16(
                    false, qf[c], false, kf.v, (short)0, s[t], false, false);
            }
        }

        // ---- online softmax (rows live on 16-lane halves; VALU permlanes) ----
        const float sc = 0.03125f;                   // 1/sqrt(1024)
#pragma unroll
        for (int r = 0; r < 8; ++r) {
            float v0 = s[0][r] * sc, v1 = s[1][r] * sc;
            float v2 = s[2][r] * sc, v3 = s[3][r] * sc;
            float mx = rowmax16(fmaxf(fmaxf(v0, v1), fmaxf(v2, v3)));
            float mnew = fmaxf(m_r[r], mx);
            float f = __expf(m_r[r] - mnew);
            float p0 = __expf(v0 - mnew), p1 = __expf(v1 - mnew);
            float p2 = __expf(v2 - mnew), p3 = __expf(v3 - mnew);
            s[0][r] = p0; s[1][r] = p1; s[2][r] = p2; s[3][r] = p3;
            float rs = rowsum16((p0 + p1) + (p2 + p3));
            l_r[r] = l_r[r] * f + rs;
            m_r[r] = mnew;
            acc_o[0][r] *= f; acc_o[1][r] *= f; acc_o[2][r] *= f; acc_o[3][r] *= f;
        }

        // ---- re-layout P (C-layout -> A-layout) through LDS ----
#pragma unroll
        for (int t = 0; t < 4; ++t)
#pragma unroll
            for (int r = 0; r < 8; ++r)
                Ps[wave][r + 8 * kh][16 * t + ln] = (bf16)s[t][r];
        __syncthreads();

        // ---- O += P V ----
#pragma unroll
        for (int c = 0; c < 2; ++c) {
            union { v16bf v; v4u q[2]; } pf;         // A: P 16 x 32 keys
            pf.q[0] = *(const v4u*)&Ps[wave][ln][32 * c + 8 * kh];
            pf.q[1] = *(const v4u*)&Ps[wave][ln][32 * c + 16 + 8 * kh];
#pragma unroll
            for (int t = 0; t < 4; ++t) {
                union { v16bf v; v4u q[2]; } vf;     // B: V[key][d16] via Vs[d][key]
                vf.q[0] = *(const v4u*)&Vs[16 * t + ln][32 * c + 8 * kh];
                vf.q[1] = *(const v4u*)&Vs[16 * t + ln][32 * c + 16 + 8 * kh];
                acc_o[t] = __builtin_amdgcn_wmma_f32_16x16x32_bf16(
                    false, pf.v, false, vf.v, (short)0, acc_o[t], false, false);
            }
        }
    }

    // ---- normalize + store to [B,S,H*64] bf16 for the output projection ----
#pragma unroll
    for (int r = 0; r < 8; ++r) {
        float inv = 1.0f / l_r[r];
        int srow = qb * 64 + 16 * wave + r + 8 * kh;
        bf16* orow = Out + ((size_t)(b * SEQ + srow)) * EMBED + h * HEAD_DIM;
#pragma unroll
        for (int t = 0; t < 4; ++t)
            orow[16 * t + ln] = (bf16)(acc_o[t][r] * inv);
    }
}

// ---------------------------------------------------------------------------
extern "C" void kernel_launch(void* const* d_in, const int* in_sizes, int n_in,
                              void* d_out, int out_size, void* d_ws, size_t ws_size,
                              hipStream_t stream) {
    const float* values = (const float*)d_in[0];
    const float* keys   = (const float*)d_in[1];
    const float* query  = (const float*)d_in[2];
    const float* Wv     = (const float*)d_in[3];
    const float* Wk     = (const float*)d_in[4];
    const float* Wq     = (const float*)d_in[5];
    const float* Wo     = (const float*)d_in[6];
    const float* bo     = (const float*)d_in[7];

    bf16* WtV = (bf16*)d_ws;
    bf16* WtK = WtV + (size_t)EMBED * EMBED;
    bf16* WtQ = WtK + (size_t)EMBED * EMBED;
    bf16* WtO = WtQ + (size_t)EMBED * EMBED;
    bf16* Vp  = WtO + (size_t)EMBED * EMBED;          // [B,H,S,64]
    bf16* Kp  = Vp  + (size_t)ROWS * EMBED;
    bf16* Qp  = Kp  + (size_t)ROWS * EMBED;
    bf16* AO  = Qp  + (size_t)ROWS * EMBED;           // [B,S,1024]

    dim3 cgrid((EMBED * EMBED) / 256);
    wt_convert_kernel<<<cgrid, 256, 0, stream>>>(Wv, WtV);
    wt_convert_kernel<<<cgrid, 256, 0, stream>>>(Wk, WtK);
    wt_convert_kernel<<<cgrid, 256, 0, stream>>>(Wq, WtQ);
    wt_convert_kernel<<<cgrid, 256, 0, stream>>>(Wo, WtO);

    dim3 ggrid(ROWS / 64, EMBED / 128);
    gemm64_kernel<true, false><<<ggrid, 128, 0, stream>>>(values, WtV, nullptr, Vp);
    gemm64_kernel<true, false><<<ggrid, 128, 0, stream>>>(keys,   WtK, nullptr, Kp);
    gemm64_kernel<true, false><<<ggrid, 128, 0, stream>>>(query,  WtQ, nullptr, Qp);

    dim3 agrid(SEQ / 64, HEADS, BATCH);
    flash_attn_kernel<<<agrid, 128, 0, stream>>>(Qp, Kp, Vp, AO);

    gemm64_kernel<false, true><<<ggrid, 128, 0, stream>>>(AO, WtO, bo, (float*)d_out);
}
